// Predict_modelv2_65704409694776
// MI455X (gfx1250) — compile-verified, matrix operand order checked by reading, not run on hardware
//
#include <hip/hip_runtime.h>
#include <hip/hip_bf16.h>
#include <climits>

// ---------------- model constants ----------------
constexpr int Nn  = 50000;
constexpr int Tt  = 5;
constexpr int Ee  = 500000;
constexpr int Fd  = 64;
constexpr int Hh  = 4;
constexpr int HF  = 256;   // H*F
constexpr int NCc = 8192;
constexpr float SLOPE = 0.2f;

typedef __attribute__((ext_vector_type(16))) _Float16 v16h;
typedef __attribute__((ext_vector_type(8)))  _Float16 v8h;
typedef __attribute__((ext_vector_type(8)))  float    v8f;

// ---------------- helpers ----------------
__device__ __forceinline__ int f2ord(float f) {
    int i = __float_as_int(f);
    return i >= 0 ? i : (i ^ 0x7FFFFFFF);
}
__device__ __forceinline__ float ord2f(int i) {
    return __int_as_float(i >= 0 ? i : (i ^ 0x7FFFFFFF));
}
__device__ __forceinline__ float lrelu(float z) { return z > 0.f ? z : SLOPE * z; }
__device__ __forceinline__ float sigmoidf(float x) { return 1.f / (1.f + __expf(-x)); }

// ---------------- WMMA GEMM: C[M,Nc] = op(A[M,K] @ B) (+bias)(+C)(relu) ----------------
// BT: B stored [Nc,K] row-major (use B^T). Otherwise B is [K,Nc] row-major.
// B is staged once per block into LDS as f16, layout [n][K] (transposed), zero-padded
// to Npad = ncTiles*16 columns. One wave32 computes one 16x16 tile with
// v_wmma_f32_16x16x32_f16; per K-step each lane does two 32-byte loads of A (f32,
// packed-converted to f16) and two 16B ds_load_b128 reads of B.
template <bool BT, bool BIAS, bool RELU, bool ACCUM>
__global__ __launch_bounds__(256)
void wmma_gemm_kernel(const float* __restrict__ A, const float* __restrict__ B,
                      const float* __restrict__ bias, float* __restrict__ C,
                      int M, int K, int Nc, int ncTiles, int totalTiles)
{
    extern __shared__ _Float16 Bl[];            // [Npad][K] f16
    const int Npad = ncTiles * 16;

    // ---- cooperative stage + f32->f16 convert of B (coalesced global reads) ----
    for (int idx = threadIdx.x; idx < Npad * K; idx += blockDim.x) {
        int n, k;
        if (BT) { n = idx / K;    k = idx - n * K; }     // global reads unit-stride in k
        else    { k = idx / Npad; n = idx - k * Npad; }  // global reads unit-stride in n
        float v = 0.f;
        if (n < Nc) v = BT ? B[(size_t)n * K + k] : B[(size_t)k * Nc + n];
        Bl[(size_t)n * K + k] = (_Float16)v;
    }
    __syncthreads();

    int wave = (blockIdx.x * blockDim.x + threadIdx.x) >> 5;
    if (wave < totalTiles) {                    // wave-uniform: EXEC all-1s inside
        int lane  = threadIdx.x & 31;
        int mt    = wave / ncTiles;
        int nt    = wave % ncTiles;
        int row   = lane & 15;
        int hi    = lane >> 4;                  // K-half select per ISA A/B layout
        int mbase = mt * 16;
        int col   = nt * 16 + row;              // < Npad always; guard vs Nc at store
        const float*    ap = A  + (size_t)(mbase + row) * K;
        const _Float16* bp = Bl + (size_t)col * K;

        v8f c = {};
        for (int k0 = 0; k0 < K; k0 += 32) {
            // A fragment: two contiguous 8-float chunks -> f16
            v8f a0 = *(const v8f*)(ap + k0 + hi * 8);
            v8f a1 = *(const v8f*)(ap + k0 + 16 + hi * 8);
            v8h ha0, ha1;
#pragma unroll
            for (int i = 0; i < 8; ++i) {
                ha0[i] = (_Float16)a0[i];
                ha1[i] = (_Float16)a1[i];
            }
            v16h a = __builtin_shufflevector(ha0, ha1, 0, 1, 2, 3, 4, 5, 6, 7,
                                             8, 9, 10, 11, 12, 13, 14, 15);
            // B fragment: two 16B LDS reads (already f16, already zero-padded)
            v8h b0 = *(const v8h*)(bp + k0 + hi * 8);
            v8h b1 = *(const v8h*)(bp + k0 + 16 + hi * 8);
            v16h b = __builtin_shufflevector(b0, b1, 0, 1, 2, 3, 4, 5, 6, 7,
                                             8, 9, 10, 11, 12, 13, 14, 15);
            c = __builtin_amdgcn_wmma_f32_16x16x32_f16(false, a, false, b,
                                                       (short)0, c, false, false);
        }

        if (col < Nc) {
            float bv = BIAS ? bias[col] : 0.f;
#pragma unroll
            for (int r = 0; r < 8; ++r) {
                int rowIdx = mbase + hi * 8 + r;   // lanes0-15: M=r ; 16-31: M=8+r
                float v = c[r] + bv;
                if (ACCUM) v += C[(size_t)rowIdx * Nc + col];
                if (RELU)  v = fmaxf(v, 0.f);
                C[(size_t)rowIdx * Nc + col] = v;
            }
        }
    }
}

// ---------------- attention scalar kernels ----------------
__global__ void alr_kernel(const float* __restrict__ h, const float* __restrict__ a_s,
                           const float* __restrict__ a_d, float* __restrict__ al,
                           float* __restrict__ ar)
{
    int i = blockIdx.x * blockDim.x + threadIdx.x;      // N*H threads
    if (i >= Nn * Hh) return;
    int n = i >> 2, hd = i & 3;
    const float* hp = h + (size_t)n * HF + hd * Fd;
    const float* as = a_s + hd * Fd;
    const float* ad = a_d + hd * Fd;
    float sa = 0.f, sd = 0.f;
#pragma unroll 8
    for (int f = 0; f < Fd; ++f) { sa += hp[f] * as[f]; sd += hp[f] * ad[f]; }
    al[i] = sa; ar[i] = sd;
}

__global__ void attn_init_kernel(float* __restrict__ acc, float* __restrict__ den,
                                 int* __restrict__ mmx)
{
    int i = blockIdx.x * blockDim.x + threadIdx.x;      // N*HF threads
    if (i >= Nn * HF) return;
    acc[i] = 0.f;
    if (i < Nn * Hh) { den[i] = 0.f; mmx[i] = INT_MIN; }
}

// atomic max of al[src] into mmx[dst] (ordered-int trick); self loops appended.
__global__ void edge_max_kernel(const int* __restrict__ src, const int* __restrict__ dst,
                                const float* __restrict__ al, int* __restrict__ mmx)
{
    int i = blockIdx.x * blockDim.x + threadIdx.x;      // (E+N)*H threads
    int Etot = Ee + Nn;
    if (i >= Etot * Hh) return;
    int e = i >> 2, hd = i & 3;
    int s = (e < Ee) ? src[e] : (e - Ee);
    int d = (e < Ee) ? dst[e] : (e - Ee);
    atomicMax(&mmx[d * Hh + hd], f2ord(al[s * Hh + hd]));
}

// one wave per edge: acc[dst,:] += exp * h[src,:], den[dst,h] += exp
__global__ __launch_bounds__(256)
void edge_agg_kernel(const int* __restrict__ src, const int* __restrict__ dst,
                     const float* __restrict__ h, const float* __restrict__ al,
                     const float* __restrict__ ar, const int* __restrict__ mmx,
                     float* __restrict__ acc, float* __restrict__ den)
{
    int Etot = Ee + Nn;
    int e = (blockIdx.x * blockDim.x + threadIdx.x) >> 5;
    if (e >= Etot) return;
    int lane = threadIdx.x & 31;
    int s = (e < Ee) ? src[e] : (e - Ee);
    int d = (e < Ee) ? dst[e] : (e - Ee);
    const float* hs = h + (size_t)s * HF;
    float* ad = acc + (size_t)d * HF;
    __builtin_prefetch(hs, 0, 3);               // global_prefetch_b8 on source row
    float ex[Hh];
#pragma unroll
    for (int hd = 0; hd < Hh; ++hd) {
        float ard = ar[d * Hh + hd];
        float ev  = lrelu(al[s * Hh + hd] + ard);
        float m   = lrelu(ord2f(mmx[d * Hh + hd]) + ard);   // segment max (monotone lrelu)
        ex[hd] = __expf(ev - m);
    }
    if (lane < Hh) atomicAdd(&den[d * Hh + lane], ex[lane]);
#pragma unroll
    for (int k = 0; k < HF / 32; ++k) {
        int f = lane + 32 * k;
        atomicAdd(&ad[f], ex[f >> 6] * hs[f]);
    }
}

__global__ void norm_bias_kernel(float* __restrict__ acc, const float* __restrict__ den,
                                 const float* __restrict__ gbias)
{
    int i = blockIdx.x * blockDim.x + threadIdx.x;      // N*HF threads
    if (i >= Nn * HF) return;
    int n = i >> 8, f = i & 255;
    acc[i] = acc[i] / (den[n * Hh + (f >> 6)] + 1e-16f) + gbias[f];
}

// ---------------- small tail kernels ----------------
__global__ void gather_kernel(const float* __restrict__ xbuf, const int* __restrict__ clf,
                              float* __restrict__ seq, int step, int half)
{
    int i = blockIdx.x * blockDim.x + threadIdx.x;      // NC*F threads
    if (i >= NCc * Fd) return;
    int nc = i >> 6, j = i & 63;
    seq[((size_t)step * NCc + nc) * (2 * Fd) + half * Fd + j] =
        xbuf[(size_t)clf[nc] * Fd + j];
}

__global__ void zero_kernel(float* __restrict__ p, int n)
{
    int i = blockIdx.x * blockDim.x + threadIdx.x;
    if (i < n) p[i] = 0.f;
}

__global__ void lstm_cell_kernel(const float* __restrict__ g, float* __restrict__ h,
                                 float* __restrict__ c, float* __restrict__ hs_out)
{
    int i = blockIdx.x * blockDim.x + threadIdx.x;      // NC*F threads
    if (i >= NCc * Fd) return;
    int n = i >> 6, j = i & 63;
    const float* gr = g + (size_t)n * 4 * Fd;           // torch gate order i,f,g,o
    float ig = sigmoidf(gr[j]);
    float fg = sigmoidf(gr[Fd + j]);
    float gg = tanhf(gr[2 * Fd + j]);
    float og = sigmoidf(gr[3 * Fd + j]);
    float cn = fg * c[i] + ig * gg;
    float hn = og * tanhf(cn);
    c[i] = cn; h[i] = hn;
    if (hs_out) hs_out[i] = hn;
}

__global__ void concat_kernel(const float* __restrict__ h1, const float* __restrict__ h2,
                              float* __restrict__ z)
{
    int i = blockIdx.x * blockDim.x + threadIdx.x;      // NC*2F threads
    if (i >= NCc * 2 * Fd) return;
    int nc = i >> 7, j = i & 127;
    z[i] = (j < Fd) ? h1[(size_t)nc * Fd + j] : h2[(size_t)nc * Fd + j - Fd];
}

// ---------------- host ----------------
#define LAUNCH_GEMM(BT, BIAS, RELU, ACCUM, A, B, BIASP, C, M, K, NC_)                    \
    do {                                                                                 \
        int nct = ((NC_) + 15) / 16;                                                     \
        int tiles = ((M) / 16) * nct;                                                    \
        int blocks = (tiles + 7) / 8;                                                    \
        size_t shb = (size_t)nct * 16 * (K) * sizeof(_Float16);                          \
        wmma_gemm_kernel<BT, BIAS, RELU, ACCUM>                                          \
            <<<blocks, 256, shb, stream>>>(A, B, BIASP, C, M, K, NC_, nct, tiles);       \
    } while (0)

static inline int cdiv(long a, long b) { return (int)((a + b - 1) / b); }

extern "C" void kernel_launch(void* const* d_in, const int* in_sizes, int n_in,
                              void* d_out, int out_size, void* d_ws, size_t ws_size,
                              hipStream_t stream)
{
    const float* emb      = (const float*)d_in[0];
    const float* gatW[2]  = {(const float*)d_in[1],  (const float*)d_in[7]};
    const float* gatAS[2] = {(const float*)d_in[2],  (const float*)d_in[8]};
    const float* gatAD[2] = {(const float*)d_in[3],  (const float*)d_in[9]};
    const float* gatB[2]  = {(const float*)d_in[4],  (const float*)d_in[10]};
    const float* linW[2]  = {(const float*)d_in[5],  (const float*)d_in[11]};
    const float* linB[2]  = {(const float*)d_in[6],  (const float*)d_in[12]};
    const float* l1Wi = (const float*)d_in[13];
    const float* l1Wh = (const float*)d_in[14];
    const float* l1b  = (const float*)d_in[15];
    const float* l2Wi = (const float*)d_in[16];
    const float* l2Wh = (const float*)d_in[17];
    const float* l2b  = (const float*)d_in[18];
    const float* fc1W = (const float*)d_in[19];
    const float* fc1b = (const float*)d_in[20];
    const float* fc2W = (const float*)d_in[21];
    const float* fc2b = (const float*)d_in[22];
    const int*   edges = (const int*)d_in[23];
    const int*   clf   = (const int*)d_in[27];
    float* out = (float*)d_out;
    (void)in_sizes; (void)n_in; (void)out_size; (void)ws_size;

    // workspace carve-up (floats)
    float* wsf = (float*)d_ws;
    size_t o = 0;
    float* hb   = wsf + o; o += (size_t)Nn * HF;          // h = x@W        [N,256]
    float* accb = wsf + o; o += (size_t)Nn * HF;          // aggregation    [N,256]
    float* xb   = wsf + o; o += (size_t)Nn * Fd;          // layer output   [N,64]
    float* alb  = wsf + o; o += (size_t)Nn * Hh;
    float* arb  = wsf + o; o += (size_t)Nn * Hh;
    int*   mmx  = (int*)(wsf + o); o += (size_t)Nn * Hh;
    float* denb = wsf + o; o += (size_t)Nn * Hh;
    float* seqb = wsf + o; o += (size_t)Tt * NCc * 2 * Fd; // LSTM input    [T,NC,128]
    float* gbuf = wsf + o; o += (size_t)NCc * 4 * Fd;      // gates         [NC,256]
    float* h1b  = wsf + o; o += (size_t)NCc * Fd;
    float* c1b  = wsf + o; o += (size_t)NCc * Fd;
    float* h2b  = wsf + o; o += (size_t)NCc * Fd;
    float* c2b  = wsf + o; o += (size_t)NCc * Fd;
    float* hsb  = wsf + o; o += (size_t)Tt * NCc * Fd;     // lstm1 outputs
    float* zb   = wsf + o; o += (size_t)NCc * 2 * Fd;      // concat(h1,h2)
    float* zzb  = wsf + o; o += (size_t)NCc * Fd;          // fc1 output

    const int TB = 256;
    const int Etot = Ee + Nn;

    // ---- GNN snapshots ----
    for (int t = 0; t < Tt; ++t) {
        const int* srcP = edges + ((size_t)t * 2 + 0) * Ee;
        const int* dstP = edges + ((size_t)t * 2 + 1) * Ee;
        for (int s = 0; s < 2; ++s) {
            const float* x = emb + (size_t)t * Nn * Fd;
            for (int l = 0; l < 2; ++l) {
                // h = x @ gW[l]   [N,64]x[64,256]
                LAUNCH_GEMM(false, false, false, false,
                            x, gatW[s] + (size_t)l * Fd * HF, nullptr, hb, Nn, Fd, HF);
                alr_kernel<<<cdiv((long)Nn * Hh, TB), TB, 0, stream>>>(
                    hb, gatAS[s] + (size_t)l * Hh * Fd, gatAD[s] + (size_t)l * Hh * Fd,
                    alb, arb);
                attn_init_kernel<<<cdiv((long)Nn * HF, TB), TB, 0, stream>>>(accb, denb, mmx);
                edge_max_kernel<<<cdiv((long)Etot * Hh, TB), TB, 0, stream>>>(
                    srcP, dstP, alb, mmx);
                edge_agg_kernel<<<cdiv((long)Etot * 32, TB), TB, 0, stream>>>(
                    srcP, dstP, hb, alb, arb, mmx, accb, denb);
                norm_bias_kernel<<<cdiv((long)Nn * HF, TB), TB, 0, stream>>>(
                    accb, denb, gatB[s] + (size_t)l * HF);
                // x = relu(acc @ lW + lb)   [N,256]x[256,64]
                LAUNCH_GEMM(false, true, true, false,
                            accb, linW[s] + (size_t)l * HF * Fd,
                            linB[s] + (size_t)l * Fd, xb, Nn, HF, Fd);
                x = xb;
            }
            // snapshots are prepended in the reference -> time axis reversed
            gather_kernel<<<cdiv((long)NCc * Fd, TB), TB, 0, stream>>>(
                xb, clf, seqb, Tt - 1 - t, s);
        }
    }

    // ---- LSTM layer 1 ----
    zero_kernel<<<cdiv((long)NCc * Fd, TB), TB, 0, stream>>>(h1b, NCc * Fd);
    zero_kernel<<<cdiv((long)NCc * Fd, TB), TB, 0, stream>>>(c1b, NCc * Fd);
    for (int t = 0; t < Tt; ++t) {
        const float* xt = seqb + (size_t)t * NCc * 2 * Fd;     // [NC,128]
        LAUNCH_GEMM(true, true, false, false, xt, l1Wi, l1b, gbuf, NCc, 2 * Fd, 4 * Fd);
        LAUNCH_GEMM(true, false, false, true, h1b, l1Wh, nullptr, gbuf, NCc, Fd, 4 * Fd);
        lstm_cell_kernel<<<cdiv((long)NCc * Fd, TB), TB, 0, stream>>>(
            gbuf, h1b, c1b, hsb + (size_t)t * NCc * Fd);
    }
    // ---- LSTM layer 2 ----
    zero_kernel<<<cdiv((long)NCc * Fd, TB), TB, 0, stream>>>(h2b, NCc * Fd);
    zero_kernel<<<cdiv((long)NCc * Fd, TB), TB, 0, stream>>>(c2b, NCc * Fd);
    for (int t = 0; t < Tt; ++t) {
        const float* xt = hsb + (size_t)t * NCc * Fd;          // [NC,64]
        LAUNCH_GEMM(true, true, false, false, xt, l2Wi, l2b, gbuf, NCc, Fd, 4 * Fd);
        LAUNCH_GEMM(true, false, false, true, h2b, l2Wh, nullptr, gbuf, NCc, Fd, 4 * Fd);
        lstm_cell_kernel<<<cdiv((long)NCc * Fd, TB), TB, 0, stream>>>(gbuf, h2b, c2b, nullptr);
    }

    // ---- classifier head ----
    concat_kernel<<<cdiv((long)NCc * 2 * Fd, TB), TB, 0, stream>>>(h1b, h2b, zb);
    LAUNCH_GEMM(false, true, true, false, zb, fc1W, fc1b, zzb, NCc, 2 * Fd, Fd);
    LAUNCH_GEMM(false, true, true, false, zzb, fc2W, fc2b, out, NCc, Fd, 2);
}
